// AG_RoPE_EncoderLayer_12910671692568
// MI455X (gfx1250) — matrix operand order repaired and checked by reading, not compiled, not run on hardware
//
#include <hip/hip_runtime.h>

// ---------------------------------------------------------------------------
// CDNA5 (gfx1250) AG-RoPE encoder layer.
// All GEMMs use v_wmma_f32_16x16x32_f16 (wave32), register-blocked for reuse.
// ---------------------------------------------------------------------------

typedef __attribute__((ext_vector_type(16))) _Float16 v16h;
typedef __attribute__((ext_vector_type(8)))  float    v8f;

union Frag16 { v16h v; uint4 q[2]; };

// A-fragment (16x32, f16, row-major source with leading dim `ld` halves).
// Layout per ISA 7.12.2: lane i (0..15) row m0+i: halves0-7 = K0..7, halves8-15 = K16..23
//                        lane i+16     row m0+i: halves0-7 = K8..15, halves8-15 = K24..31
__device__ __forceinline__ v16h load_a_frag(const _Float16* base, int ld, int m0, int k0) {
    int lane = threadIdx.x & 31;
    int i = lane & 15, hi = lane >> 4;
    const _Float16* p = base + (size_t)(m0 + i) * ld + k0 + hi * 8;
    Frag16 f;
    f.q[0] = *(const uint4*)(p);
    f.q[1] = *(const uint4*)(p + 16);
    return f.v;
}

// B-fragment (32x16, f16). Source stored N-major: row n holds K contiguous ("W[o][c]").
// lane j (0..15) col n0+j: halves0-7 = K0..7, halves8-15 = K16..23 ; lane j+16: halves0-7 = K8..15, halves8-15 = K24..31
__device__ __forceinline__ v16h load_b_frag(const _Float16* base, int ld, int n0, int k0) {
    int lane = threadIdx.x & 31;
    int j = lane & 15, hi = lane >> 4;
    const _Float16* p = base + (size_t)(n0 + j) * ld + k0 + hi * 8;
    Frag16 f;
    f.q[0] = *(const uint4*)(p);
    f.q[1] = *(const uint4*)(p + 16);
    return f.v;
}

__device__ __forceinline__ v8f wmma_f16(v16h a, v16h b, v8f c) {
    return __builtin_amdgcn_wmma_f32_16x16x32_f16(false, a, false, b, (short)0, c, false, false);
}

// ---------------------------------------------------------------------------
// Workspace layout (bytes)
// ---------------------------------------------------------------------------
constexpr size_t OFF_WQ   = 0;                       // 256*256 f16
constexpr size_t OFF_WK   = OFF_WQ   + 131072;
constexpr size_t OFF_WV   = OFF_WK   + 131072;
constexpr size_t OFF_WM   = OFF_WV   + 131072;
constexpr size_t OFF_W1   = OFF_WM   + 131072;       // 512*512 f16
constexpr size_t OFF_W2   = OFF_W1   + 524288;       // 256*512 f16
constexpr size_t OFF_QRAW = OFF_W2   + 262144;       // 8192*256 f32
constexpr size_t OFF_SRAW = OFF_QRAW + 8388608;      // 8192*256 f32
constexpr size_t OFF_QTOK = OFF_SRAW + 8388608;      // 8192*256 f16
constexpr size_t OFF_STOK = OFF_QTOK + 4194304;      // 8192*256 f16
constexpr size_t OFF_SIN  = OFF_STOK + 4194304;      // 1024*256 f32
constexpr size_t OFF_COS  = OFF_SIN  + 1048576;
constexpr size_t OFF_QR   = OFF_COS  + 1048576;      // [8][8][1024][32] f16
constexpr size_t OFF_KR   = OFF_QR   + 4194304;
constexpr size_t OFF_VT   = OFF_KR   + 4194304;      // [8][8][32][1024] f16
constexpr size_t OFF_O16  = OFF_VT   + 4194304;      // 8192*256 f16
constexpr size_t OFF_MTOK = OFF_O16  + 4194304;      // 8192*256 f32

// ---------------------------------------------------------------------------
// K1: f32 -> f16 weight conversion
// ---------------------------------------------------------------------------
__global__ void k_cvt(const float* __restrict__ s, _Float16* __restrict__ d, int n) {
    int i = blockIdx.x * 256 + threadIdx.x;
    if (i < n) d[i] = (_Float16)s[i];
}

// ---------------------------------------------------------------------------
// K2: depthwise conv 4x4 stride4 (x) + maxpool 4x4 (source), NCHW -> token-major
// grid = 8*256 blocks (n,c), 256 threads
// ---------------------------------------------------------------------------
__global__ void k_agg(const float* __restrict__ x, const float* __restrict__ src,
                      const float* __restrict__ wagg,
                      float* __restrict__ qraw, float* __restrict__ sraw) {
    int c = blockIdx.x & 255, n = blockIdx.x >> 8;
    float w[16];
#pragma unroll
    for (int i = 0; i < 16; ++i) w[i] = wagg[c * 16 + i];
    int tid = threadIdx.x;
    const float* xb0 = x   + ((size_t)(n * 256 + c)) * 16384;
    const float* sb0 = src + ((size_t)(n * 256 + c)) * 16384;
    for (int rep = 0; rep < 4; ++rep) {
        int Y = rep * 8 + (tid >> 5), X = tid & 31;
        const float* xb = xb0 + (Y * 4) * 128 + X * 4;
        const float* sb = sb0 + (Y * 4) * 128 + X * 4;
        float acc = 0.f, mx = -3.4e38f;
#pragma unroll
        for (int dy = 0; dy < 4; ++dy) {
            float4 xv = *(const float4*)(xb + dy * 128);
            float4 sv = *(const float4*)(sb + dy * 128);
            acc += xv.x * w[dy*4] + xv.y * w[dy*4+1] + xv.z * w[dy*4+2] + xv.w * w[dy*4+3];
            mx = fmaxf(mx, fmaxf(fmaxf(sv.x, sv.y), fmaxf(sv.z, sv.w)));
        }
        int tok = n * 1024 + Y * 32 + X;
        qraw[(size_t)tok * 256 + c] = acc;
        sraw[(size_t)tok * 256 + c] = mx;
    }
}

// ---------------------------------------------------------------------------
// K3: LayerNorm over C=256, f32 -> f16 tokens.  grid = 8192, 256 threads
// ---------------------------------------------------------------------------
__global__ void k_ln(const float* __restrict__ qraw, const float* __restrict__ sraw,
                     const float* __restrict__ g, const float* __restrict__ b,
                     _Float16* __restrict__ qt, _Float16* __restrict__ st) {
    __shared__ float red[256];
    int tok = blockIdx.x, c = threadIdx.x;
    for (int which = 0; which < 2; ++which) {
        const float* s0 = which ? sraw : qraw;
        _Float16* dst   = which ? st : qt;
        float v = s0[(size_t)tok * 256 + c];
        red[c] = v; __syncthreads();
        for (int s = 128; s > 0; s >>= 1) { if (c < s) red[c] += red[c + s]; __syncthreads(); }
        float mu = red[0] * (1.f / 256.f); __syncthreads();
        float d = v - mu;
        red[c] = d * d; __syncthreads();
        for (int s = 128; s > 0; s >>= 1) { if (c < s) red[c] += red[c + s]; __syncthreads(); }
        float rs = rsqrtf(red[0] * (1.f / 256.f) + 1e-5f); __syncthreads();
        dst[(size_t)tok * 256 + c] = (_Float16)(d * rs * g[c] + b[c]);
    }
}

// ---------------------------------------------------------------------------
// K4: RoPE tables (h=w=32, C=256, dq=64).  grid = 1024, 256 threads
// ---------------------------------------------------------------------------
__global__ void k_rope(float* __restrict__ sinT, float* __restrict__ cosT) {
    int l = blockIdx.x, c = threadIdx.x;
    int y = l >> 5, xc = l & 31;
    int p = c >> 1;
    int t = p >> 1;
    float pos = (p & 1) ? (float)(xc + 1) : (float)(y + 1);
    float div = __expf(-(float)t * (9.210340371976184f / 64.f)); // ln(10000)/dq
    float a = pos * div;
    sinT[l * 256 + c] = __sinf(a);
    cosT[l * 256 + c] = __cosf(a);
}

// ---------------------------------------------------------------------------
// K5: Q/K/V projection GEMM (X[8192,256] x W^T) with fused RoPE epilogue.
// grid = (512, 4, 3), block = 32. Register-blocked: one A-frag reused across
// 4 N-tiles per k-step (4x cut in A traffic).
// Q -> Qr[n][h][l][32], K -> Kr[n][h][l][32], V -> Vt[n][h][32][l]
// ---------------------------------------------------------------------------
__global__ void k_qkv(const _Float16* __restrict__ qtok, const _Float16* __restrict__ stok,
                      const _Float16* __restrict__ wq, const _Float16* __restrict__ wk,
                      const _Float16* __restrict__ wv,
                      const float* __restrict__ sinT, const float* __restrict__ cosT,
                      _Float16* __restrict__ Qr, _Float16* __restrict__ Kr,
                      _Float16* __restrict__ Vt) {
    int mt = blockIdx.x, ntb = blockIdx.y * 4, mat = blockIdx.z;
    const _Float16* A = (mat == 0) ? qtok : stok;
    const _Float16* W = (mat == 0) ? wq : (mat == 1 ? wk : wv);
    __builtin_prefetch(W + (size_t)ntb * 16 * 256, 0, 1);
    v8f acc[4] = {};
    for (int ks = 0; ks < 8; ++ks) {
        v16h a = load_a_frag(A, 256, mt * 16, ks * 32);
#pragma unroll
        for (int u = 0; u < 4; ++u) {
            v16h b = load_b_frag(W, 256, (ntb + u) * 16, ks * 32);
            acc[u] = wmma_f16(a, b, acc[u]);
        }
    }
    int lane = threadIdx.x, j = lane & 15, hi = lane >> 4;
    if (mat < 2) {
        _Float16* dst = (mat == 0) ? Qr : Kr;
        float rsign = (j & 1) ? 1.f : -1.f;    // rotate_half pairing (col parity == j parity)
#pragma unroll
        for (int u = 0; u < 4; ++u) {
            int col = (ntb + u) * 16 + j;
            int hh = col >> 5, dd = col & 31;
#pragma unroll
            for (int r = 0; r < 8; ++r) {
                float v = acc[u][r];
                float pv = __shfl_xor(v, 1, 32);   // partner channel (col^1), same row
                int tok = mt * 16 + hi * 8 + r;
                int n = tok >> 10, l = tok & 1023;
                float sv = sinT[l * 256 + col], cv = cosT[l * 256 + col];
                float o = v * cv + rsign * pv * sv;
                dst[((size_t)((n * 8 + hh) * 1024 + l)) * 32 + dd] = (_Float16)o;
            }
        }
    } else {
#pragma unroll
        for (int u = 0; u < 4; ++u) {
            int col = (ntb + u) * 16 + j;
            int hh = col >> 5, dd = col & 31;
#pragma unroll
            for (int r = 0; r < 8; ++r) {
                int tok = mt * 16 + hi * 8 + r;
                int n = tok >> 10, l = tok & 1023;
                Vt[((size_t)((n * 8 + hh) * 32 + dd)) * 1024 + l] = (_Float16)acc[u][r];
            }
        }
    }
}

// ---------------------------------------------------------------------------
// K6: attention. grid = (64 l-blocks, 8 heads, 8 batch), block = 32 (1 wave).
// 64KB LDS logit row-block (16x1024 f32); exact softmax; in-place f16 P alias.
// ---------------------------------------------------------------------------
__global__ void k_attn(const _Float16* __restrict__ Qr, const _Float16* __restrict__ Kr,
                       const _Float16* __restrict__ Vt, _Float16* __restrict__ O16) {
    __shared__ float logits[16 * 1024];
    int lb = blockIdx.x, h = blockIdx.y, n = blockIdx.z;
    int lane = threadIdx.x, j = lane & 15, hi = lane >> 4;
    const _Float16* Qb = Qr + ((size_t)(n * 8 + h)) * 1024 * 32;
    const _Float16* Kb = Kr + ((size_t)(n * 8 + h)) * 1024 * 32;
    const _Float16* Vb = Vt + ((size_t)(n * 8 + h)) * 32 * 1024;

    v16h aq = load_a_frag(Qb, 32, lb * 16, 0);     // Q block 16x32 (K = d = 32)
    const float scale = 0.17677669529663687f;      // 1/sqrt(32)
    for (int sb = 0; sb < 64; ++sb) {
        v16h b = load_b_frag(Kb, 32, sb * 16, 0);  // K^T block 32x16
        v8f c = {};
        c = wmma_f16(aq, b, c);
#pragma unroll
        for (int r = 0; r < 8; ++r)
            logits[(hi * 8 + r) * 1024 + sb * 16 + j] = c[r] * scale;
    }
    __syncthreads();

    _Float16* P = (_Float16*)logits;               // f16 rows alias f32 rows (pitch 2048 halves)
    if (lane < 16) {
        float* lr = logits + lane * 1024;
        _Float16* pr = P + lane * 2048;
        float mx = -3.4e38f;
        for (int s = 0; s < 1024; ++s) mx = fmaxf(mx, lr[s]);
        float sum = 0.f;
        for (int s = 0; s < 1024; ++s) sum += __expf(lr[s] - mx);
        float inv = 1.f / sum;
        for (int s = 0; s < 1024; ++s) {           // ascending: f16 write never clobbers unread f32
            float e = __expf(lr[s] - mx) * inv;
            pr[s] = (_Float16)e;
        }
    }
    __syncthreads();

#pragma unroll
    for (int nd = 0; nd < 2; ++nd) {               // output d split 2x16
        v8f acc = {};
        for (int kb = 0; kb < 32; ++kb) {
            v16h a = load_a_frag(P, 2048, 0, kb * 32);
            v16h b = load_b_frag(Vb, 1024, nd * 16, kb * 32);   // Vt rows are d, contiguous s
            acc = wmma_f16(a, b, acc);
        }
#pragma unroll
        for (int r = 0; r < 8; ++r) {
            int tok = n * 1024 + lb * 16 + hi * 8 + r;
            O16[(size_t)tok * 256 + h * 32 + nd * 16 + j] = (_Float16)acc[r];
        }
    }
}

// ---------------------------------------------------------------------------
// K7: merge GEMM  O[8192,256] x w_merge^T -> m_tok f32
// grid = (512, 4), block = 32, register-blocked 1Mx4N.
// ---------------------------------------------------------------------------
__global__ void k_merge(const _Float16* __restrict__ O, const _Float16* __restrict__ W,
                        float* __restrict__ M) {
    int mt = blockIdx.x, ntb = blockIdx.y * 4;
    v8f acc[4] = {};
    for (int ks = 0; ks < 8; ++ks) {
        v16h a = load_a_frag(O, 256, mt * 16, ks * 32);
#pragma unroll
        for (int u = 0; u < 4; ++u) {
            v16h b = load_b_frag(W, 256, (ntb + u) * 16, ks * 32);
            acc[u] = wmma_f16(a, b, acc[u]);
        }
    }
    int lane = threadIdx.x, j = lane & 15, hi = lane >> 4;
#pragma unroll
    for (int u = 0; u < 4; ++u)
#pragma unroll
        for (int r = 0; r < 8; ++r)
            M[(size_t)(mt * 16 + hi * 8 + r) * 256 + (ntb + u) * 16 + j] = acc[u][r];
}

// ---------------------------------------------------------------------------
// K8: fused bilinear upsample + concat + MLP + LN + residual.
// grid = 131072/64 blocks of 64 tokens, 256 threads (8 waves).
// 128KB *dynamic* LDS (cat 64KB + hid 64KB; out f32 aliases cat).
// Register blocking: each B-fragment (weight) loaded once per k-step and
// reused across all 4 M-tiles -> W1 traffic ~1GB total, W2 ~0.5GB (L2-resident).
// ---------------------------------------------------------------------------
__global__ void k_mlp(const float* __restrict__ x, const float* __restrict__ m_tok,
                      const _Float16* __restrict__ W1, const _Float16* __restrict__ W2,
                      const float* __restrict__ g2, const float* __restrict__ b2,
                      float* __restrict__ out) {
    extern __shared__ char smem[];
    _Float16* catb = (_Float16*)smem;              // [64][512] f16 = 64KB
    _Float16* hid  = (_Float16*)(smem + 65536);    // [64][512] f16 = 64KB
    float*    outb = (float*)smem;                 // [64][256] f32 = 64KB (aliases cat)

    int tid = threadIdx.x;
    int tok0 = blockIdx.x * 64;
    int n = tok0 >> 14;
    int pix0 = tok0 & 16383;
    const float* mb = m_tok + (size_t)n * 1024 * 256;

    __builtin_prefetch(W1 + (size_t)tid * 64, 0, 1);   // warm first W1 slab in L2/WGP$

    // --- stage cat[:,0:256] = x (coalesced NCHW read, LDS transpose) ---
    for (int idx = tid; idx < 64 * 256; idx += 256) {
        int c = idx >> 6, p = idx & 63;
        catb[p * 512 + c] = (_Float16)x[((size_t)(n * 256 + c)) * 16384 + pix0 + p];
    }
    // --- stage cat[:,256:512] = bilinear-upsampled m (half-pixel, edge clamp) ---
    for (int idx = tid; idx < 64 * 256; idx += 256) {
        int c = idx >> 6, p = idx & 63;
        int pix = pix0 + p; int Y = pix >> 7, X = pix & 127;
        float sy = (Y + 0.5f) * 0.25f - 0.5f; float fy = floorf(sy); int y0 = (int)fy; float wy = sy - fy;
        float sx = (X + 0.5f) * 0.25f - 0.5f; float fx = floorf(sx); int x0 = (int)fx; float wx = sx - fx;
        int y0c = y0 < 0 ? 0 : (y0 > 31 ? 31 : y0);
        int y1c = (y0 + 1) < 0 ? 0 : ((y0 + 1) > 31 ? 31 : (y0 + 1));
        int x0c = x0 < 0 ? 0 : (x0 > 31 ? 31 : x0);
        int x1c = (x0 + 1) < 0 ? 0 : ((x0 + 1) > 31 ? 31 : (x0 + 1));
        float v00 = mb[(size_t)(y0c * 32 + x0c) * 256 + c];
        float v01 = mb[(size_t)(y0c * 32 + x1c) * 256 + c];
        float v10 = mb[(size_t)(y1c * 32 + x0c) * 256 + c];
        float v11 = mb[(size_t)(y1c * 32 + x1c) * 256 + c];
        float v = (1.f - wy) * ((1.f - wx) * v00 + wx * v01)
                +        wy  * ((1.f - wx) * v10 + wx * v11);
        catb[p * 512 + 256 + c] = (_Float16)v;
    }
    __syncthreads();

    int wave = tid >> 5;
    int lane = tid & 31, j = lane & 15, hi = lane >> 4;

    // --- GEMM1: hid[64,512] = LeakyReLU(cat[64,512] x W1^T) ---
    // wave handles nt = wave*4 .. wave*4+3, all 4 m-tiles; B loaded once per (nt,ks).
    {
        v8f acc[4][4] = {};                         // [nv][mt]
        for (int ks = 0; ks < 16; ++ks) {
            v16h a[4];
#pragma unroll
            for (int mt = 0; mt < 4; ++mt)
                a[mt] = load_a_frag(catb, 512, mt * 16, ks * 32);
#pragma unroll
            for (int nv = 0; nv < 4; ++nv) {
                int nt = wave * 4 + nv;
                v16h b = load_b_frag(W1, 512, nt * 16, ks * 32);
#pragma unroll
                for (int mt = 0; mt < 4; ++mt)
                    acc[nv][mt] = wmma_f16(a[mt], b, acc[nv][mt]);
            }
        }
#pragma unroll
        for (int nv = 0; nv < 4; ++nv) {
            int nt = wave * 4 + nv;
#pragma unroll
            for (int mt = 0; mt < 4; ++mt)
#pragma unroll
                for (int r = 0; r < 8; ++r) {
                    float v = acc[nv][mt][r];
                    v = (v >= 0.f) ? v : 0.01f * v;          // LeakyReLU(0.01)
                    hid[(mt * 16 + hi * 8 + r) * 512 + nt * 16 + j] = (_Float16)v;
                }
        }
    }
    __syncthreads();

    // --- GEMM2: out[64,256] = hid x W2^T (f32 into cat region) ---
    {
        v8f acc[2][4] = {};                         // [nv][mt]
        for (int ks = 0; ks < 16; ++ks) {
            v16h a[4];
#pragma unroll
            for (int mt = 0; mt < 4; ++mt)
                a[mt] = load_a_frag(hid, 512, mt * 16, ks * 32);
#pragma unroll
            for (int nv = 0; nv < 2; ++nv) {
                int nt = wave * 2 + nv;
                v16h b = load_b_frag(W2, 512, nt * 16, ks * 32);
#pragma unroll
                for (int mt = 0; mt < 4; ++mt)
                    acc[nv][mt] = wmma_f16(a[mt], b, acc[nv][mt]);
            }
        }
        __syncthreads();                            // cat reads done everywhere before alias write
#pragma unroll
        for (int nv = 0; nv < 2; ++nv) {
            int nt = wave * 2 + nv;
#pragma unroll
            for (int mt = 0; mt < 4; ++mt)
#pragma unroll
                for (int r = 0; r < 8; ++r)
                    outb[(mt * 16 + hi * 8 + r) * 256 + nt * 16 + j] = acc[nv][mt][r];
        }
    }
    __syncthreads();

    // --- LayerNorm per token (4 threads per row, shfl reduce over masks 1,2) ---
    {
        int row = tid >> 2, q = tid & 3;
        float s = 0.f, ss = 0.f;
        for (int c = q * 64; c < q * 64 + 64; ++c) {
            float v = outb[row * 256 + c]; s += v; ss += v * v;
        }
        s += __shfl_xor(s, 1, 32); ss += __shfl_xor(ss, 1, 32);
        s += __shfl_xor(s, 2, 32); ss += __shfl_xor(ss, 2, 32);
        float mu = s * (1.f / 256.f);
        float var = ss * (1.f / 256.f) - mu * mu;
        float rs = rsqrtf(var + 1e-5f);
        for (int c = q * 64; c < q * 64 + 64; ++c) {
            float v = outb[row * 256 + c];
            outb[row * 256 + c] = (v - mu) * rs * g2[c] + b2[c];
        }
    }
    __syncthreads();

    // --- residual add + coalesced NCHW store (64-wide contiguous runs) ---
    for (int it = 0; it < 64; ++it) {
        int c = it * 4 + (tid >> 6);
        int p = tid & 63;
        size_t addr = ((size_t)(n * 256 + c)) * 16384 + pix0 + p;
        out[addr] = outb[p * 256 + c] + x[addr];
    }
}

// ---------------------------------------------------------------------------
// host launch
// ---------------------------------------------------------------------------
extern "C" void kernel_launch(void* const* d_in, const int* in_sizes, int n_in,
                              void* d_out, int out_size, void* d_ws, size_t ws_size,
                              hipStream_t stream) {
    (void)in_sizes; (void)n_in; (void)out_size; (void)ws_size;
    const float* x      = (const float*)d_in[0];
    const float* source = (const float*)d_in[1];
    const float* w_agg  = (const float*)d_in[2];
    const float* w_q    = (const float*)d_in[3];
    const float* w_k    = (const float*)d_in[4];
    const float* w_v    = (const float*)d_in[5];
    const float* w_mrg  = (const float*)d_in[6];
    const float* w_mlp1 = (const float*)d_in[7];
    const float* w_mlp2 = (const float*)d_in[8];
    const float* g1     = (const float*)d_in[9];
    const float* b1     = (const float*)d_in[10];
    const float* g2     = (const float*)d_in[11];
    const float* b2     = (const float*)d_in[12];
    float* out = (float*)d_out;

    char* ws = (char*)d_ws;
    _Float16* wq16 = (_Float16*)(ws + OFF_WQ);
    _Float16* wk16 = (_Float16*)(ws + OFF_WK);
    _Float16* wv16 = (_Float16*)(ws + OFF_WV);
    _Float16* wm16 = (_Float16*)(ws + OFF_WM);
    _Float16* w116 = (_Float16*)(ws + OFF_W1);
    _Float16* w216 = (_Float16*)(ws + OFF_W2);
    float*    qraw = (float*)(ws + OFF_QRAW);
    float*    sraw = (float*)(ws + OFF_SRAW);
    _Float16* qtok = (_Float16*)(ws + OFF_QTOK);
    _Float16* stok = (_Float16*)(ws + OFF_STOK);
    float*    sinT = (float*)(ws + OFF_SIN);
    float*    cosT = (float*)(ws + OFF_COS);
    _Float16* Qr   = (_Float16*)(ws + OFF_QR);
    _Float16* Kr   = (_Float16*)(ws + OFF_KR);
    _Float16* Vt   = (_Float16*)(ws + OFF_VT);
    _Float16* O16  = (_Float16*)(ws + OFF_O16);
    float*    mtok = (float*)(ws + OFF_MTOK);

    // weight conversion (f32 -> f16)
    k_cvt<<<256, 256, 0, stream>>>(w_q,    wq16, 65536);
    k_cvt<<<256, 256, 0, stream>>>(w_k,    wk16, 65536);
    k_cvt<<<256, 256, 0, stream>>>(w_v,    wv16, 65536);
    k_cvt<<<256, 256, 0, stream>>>(w_mrg,  wm16, 65536);
    k_cvt<<<1024, 256, 0, stream>>>(w_mlp1, w116, 262144);
    k_cvt<<<512, 256, 0, stream>>>(w_mlp2, w216, 131072);

    // aggregation + LN + RoPE tables
    k_agg<<<8 * 256, 256, 0, stream>>>(x, source, w_agg, qraw, sraw);
    k_ln<<<8192, 256, 0, stream>>>(qraw, sraw, g1, b1, qtok, stok);
    k_rope<<<1024, 256, 0, stream>>>(sinT, cosT);

    // Q/K/V projections with fused RoPE (register-blocked 1Mx4N per wave)
    k_qkv<<<dim3(512, 4, 3), 32, 0, stream>>>(qtok, stok, wq16, wk16, wv16,
                                              sinT, cosT, Qr, Kr, Vt);
    // attention
    k_attn<<<dim3(64, 8, 8), 32, 0, stream>>>(Qr, Kr, Vt, O16);
    // merge projection (register-blocked)
    k_merge<<<dim3(512, 4), 32, 0, stream>>>(O16, wm16, mtok);
    // upsample + concat + MLP + LN + residual (64-token tiles, 128KB dynamic LDS)
    k_mlp<<<131072 / 64, 256, 131072, stream>>>(x, mtok, w116, w216, g2, b2, out);
}
